// Attn_Head_20418274525779
// MI455X (gfx1250) — matrix operand order repaired and checked by reading, not compiled, not run on hardware
//
#include <hip/hip_runtime.h>
#include <hip/hip_bf16.h>
#include <math.h>

// Problem sizes (fixed by the reference)
constexpr int NN   = 8192;   // nodes
constexpr int INCH = 256;    // input channels
constexpr int OSZ  = 64;     // output features

typedef __attribute__((ext_vector_type(16))) __bf16 v16bf;
typedef __attribute__((ext_vector_type(8)))  float  v8f;

#define LOG2E 1.44269504088896340736f

__device__ __forceinline__ float fast_exp(float x) {
    return __builtin_amdgcn_exp2f(x * LOG2E);
}

// attention numerator without the max-shift: exp(leaky_relu(f1[j]+f2[i]) + bias)
// (the exp(-m[j])/denom[j] factor is pre-folded into the B matrix sf')
__device__ __forceinline__ float coef1(float f1v, float f2v, float bv) {
    float u  = f1v + f2v;
    float lr = fmaxf(u, 0.01f * u);          // leaky_relu, slope 0.01
    return __builtin_amdgcn_exp2f((lr + bv) * LOG2E);
}

template <int BASE>
__device__ __forceinline__ void pack4(v16bf& a, float4 f, float4 b, float f2i) {
    a[BASE + 0] = (__bf16)coef1(f.x, f2i, b.x);
    a[BASE + 1] = (__bf16)coef1(f.y, f2i, b.y);
    a[BASE + 2] = (__bf16)coef1(f.z, f2i, b.z);
    a[BASE + 3] = (__bf16)coef1(f.w, f2i, b.w);
}

// ---------------------------------------------------------------------------
// K1: seq_fts[r][c] = dot(x[r,:], W1[c,:])   (fp32, tiny GEMM, 0.27 GFLOP)
// ---------------------------------------------------------------------------
__global__ void proj_kernel(const float* __restrict__ x,
                            const float* __restrict__ W1,
                            float* __restrict__ seq) {
    const int tid = threadIdx.x;
    const int r = blockIdx.x * 4 + (tid >> 6);
    const int c = tid & 63;
    const float4* xr = (const float4*)(x  + (size_t)r * INCH);
    const float4* wr = (const float4*)(W1 + (size_t)c * INCH);
    float acc = 0.f;
#pragma unroll 8
    for (int k = 0; k < INCH / 4; ++k) {
        float4 a = xr[k], b = wr[k];
        acc += a.x * b.x + a.y * b.y + a.z * b.z + a.w * b.w;
    }
    seq[(size_t)r * OSZ + c] = acc;
}

// ---------------------------------------------------------------------------
// K2: f1[r] = seq[r,:]·a1 ; f2[r] = seq[r,:]·a2
// ---------------------------------------------------------------------------
__global__ void fvec_kernel(const float* __restrict__ seq,
                            const float* __restrict__ a1,
                            const float* __restrict__ a2,
                            float* __restrict__ f1,
                            float* __restrict__ f2) {
    const int r = blockIdx.x * blockDim.x + threadIdx.x;
    const float4* sr = (const float4*)(seq + (size_t)r * OSZ);
    const float4* p1 = (const float4*)a1;
    const float4* p2 = (const float4*)a2;
    float s1 = 0.f, s2 = 0.f;
#pragma unroll
    for (int k = 0; k < OSZ / 4; ++k) {
        float4 v = sr[k], b1 = p1[k], b2 = p2[k];
        s1 += v.x * b1.x + v.y * b1.y + v.z * b1.z + v.w * b1.w;
        s2 += v.x * b2.x + v.y * b2.y + v.z * b2.z + v.w * b2.w;
    }
    f1[r] = s1;
    f2[r] = s2;
}

// ---------------------------------------------------------------------------
// K3: online softmax stats per column j (softmax over axis 0 == rows i).
// Emits combined scale s[j] = exp(-m[j]) / sum_i exp(l[i,j]-m[j]).
// First full streamed read of bias_mx (256 MB), coalesced 1 KB/row per block.
// ---------------------------------------------------------------------------
__global__ void colstat_kernel(const float* __restrict__ bias,
                               const float* __restrict__ f1,
                               const float* __restrict__ f2,
                               float* __restrict__ scaleOut) {
    const int j = blockIdx.x * blockDim.x + threadIdx.x;
    const float f1j = f1[j];
    float m = -3.0e38f, sum = 0.f;
    for (int i = 0; i < NN; ++i) {
        float u  = f1j + f2[i];
        float lr = fmaxf(u, 0.01f * u);
        float l  = lr + bias[(size_t)i * NN + j];
        if (l <= m) {                      // common path: one transcendental
            sum += fast_exp(l - m);
        } else {                           // rare max update
            sum = sum * fast_exp(m - l) + 1.0f;
            m = l;
        }
    }
    scaleOut[j] = fast_exp(-m) / sum;      // exp(-m[j]) / denom[j]
}

// ---------------------------------------------------------------------------
// K4: sf'[j,k] = bf16( seq[j,k] * s[j] )   (1 MB, stays L2-resident)
// ---------------------------------------------------------------------------
__global__ void scale_kernel(const float* __restrict__ seq,
                             const float* __restrict__ scl,
                             __bf16* __restrict__ sfb) {
    const int gid = blockIdx.x * blockDim.x + threadIdx.x;
    const int j = gid >> 6;
    sfb[gid] = (__bf16)(seq[gid] * scl[j]);
}

// ---------------------------------------------------------------------------
// K5: ret[i,k] = sum_j exp(lrelu(f1[j]+f2[i]) + bias[i,j]) * sf'[j,k], then ELU.
// One 16-row output block per workgroup; 8 waves split j (1024 each);
// WMMA bf16 16x16x32; software-pipelined stage regs (pinned with a scheduling
// barrier so the compiler cannot sink the B-tile loads back onto the WMMAs)
// + deep speculative bias prefetch.  Second full read of bias_mx.
// ---------------------------------------------------------------------------

// Load one pipeline stage: f1 (L0-hot), bias (HBM stream), sf' B-tiles (L2-hot)
#define LOAD_STAGE(J, Fa, Fb, Fc_, Fd, Ba, Bb, Bc_, Bd, T0, T1, T2, T3)      \
    {                                                                        \
        const int o0 = (J) + hv * 8;                                         \
        const int o1 = (J) + 16 + hv * 8;                                    \
        Fa  = *(const float4*)(f1 + o0);                                     \
        Fb  = *(const float4*)(f1 + o0 + 4);                                 \
        Fc_ = *(const float4*)(f1 + o1);                                     \
        Fd  = *(const float4*)(f1 + o1 + 4);                                 \
        Ba  = *(const float4*)(biasrow + o0);                                \
        Bb  = *(const float4*)(biasrow + o0 + 4);                            \
        Bc_ = *(const float4*)(biasrow + o1);                                \
        Bd  = *(const float4*)(biasrow + o1 + 4);                            \
        _Pragma("unroll")                                                    \
        for (int s_ = 0; s_ < 16; ++s_) {                                    \
            const int kk_ = (s_ < 8) ? (o0 + s_) : (o1 + s_ - 8);            \
            const __bf16* bp_ = sfb + (size_t)kk_ * OSZ + mrow;              \
            T0[s_] = bp_[0]; T1[s_] = bp_[16];                               \
            T2[s_] = bp_[32]; T3[s_] = bp_[48];                              \
        }                                                                    \
    }

__global__ void __launch_bounds__(256) attn_kernel(
        const float* __restrict__ bias,
        const float* __restrict__ f1,
        const float* __restrict__ f2,
        const __bf16* __restrict__ sfb,
        float* __restrict__ out) {
    __shared__ float red[8 * 1024];            // 8 waves x (16 x 64) partials

    const int tid  = threadIdx.x;
    const int wave = tid >> 5;
    const int lane = tid & 31;
    const int hv   = lane >> 4;                // lane half (ISA 16-bit A layout)
    const int mrow = lane & 15;                // A row; B/D column within tile
    const int i0   = blockIdx.x * 16;
    const int i    = i0 + mrow;

    const float  f2i     = f2[i];
    const float* biasrow = bias + (size_t)i * NN;

    v8f acc0 = {}, acc1 = {}, acc2 = {}, acc3 = {};

    const int jbeg = wave * 1024;
    const int jend = jbeg + 1024;

    // pipeline prologue
    float4 Fc0, Fc1, Fc2, Fc3, Bc0, Bc1, Bc2, Bc3;
    v16bf  bt0, bt1, bt2, bt3;
    LOAD_STAGE(jbeg, Fc0, Fc1, Fc2, Fc3, Bc0, Bc1, Bc2, Bc3, bt0, bt1, bt2, bt3);

#pragma unroll 2
    for (int j0 = jbeg; j0 < jend; j0 += 32) {
        const int jn = (j0 + 32 < jend) ? (j0 + 32) : jbeg;

        // deep speculative prefetch of the HBM bias stream (4-5 chunks ahead;
        // 32 lanes x 64B lines cover one full future 128B chunk per instr;
        // running past the buffer tail is dropped silently by hardware)
        __builtin_prefetch(biasrow + j0 + 128 + hv * 16, 0, 0);
        __builtin_prefetch(biasrow + j0 + 160 + hv * 16, 0, 0);

        // issue next stage's loads before consuming the current stage
        float4 Fn0, Fn1, Fn2, Fn3, Bn0, Bn1, Bn2, Bn3;
        v16bf  nt0, nt1, nt2, nt3;
        LOAD_STAGE(jn, Fn0, Fn1, Fn2, Fn3, Bn0, Bn1, Bn2, Bn3, nt0, nt1, nt2, nt3);

        // Pin all next-stage loads above this point: without it the scheduler
        // sinks the B-tile loads back to just before each WMMA, turning every
        // WMMA into an s_wait_loadcnt 0 full-latency stall.
        __builtin_amdgcn_sched_barrier(0);

        // A tile from current stage: exp(lrelu(f1[j]+f2[i]) + bias[i,j]) -> bf16
        v16bf a;
        pack4<0>(a, Fc0, Bc0, f2i);
        pack4<4>(a, Fc1, Bc1, f2i);
        pack4<8>(a, Fc2, Bc2, f2i);
        pack4<12>(a, Fc3, Bc3, f2i);

        acc0 = __builtin_amdgcn_wmma_f32_16x16x32_bf16(false, a, false, bt0, (short)0, acc0, false, false);
        acc1 = __builtin_amdgcn_wmma_f32_16x16x32_bf16(false, a, false, bt1, (short)0, acc1, false, false);
        acc2 = __builtin_amdgcn_wmma_f32_16x16x32_bf16(false, a, false, bt2, (short)0, acc2, false, false);
        acc3 = __builtin_amdgcn_wmma_f32_16x16x32_bf16(false, a, false, bt3, (short)0, acc3, false, false);

        // rotate pipeline (copies eliminated by the 2x unroll)
        Fc0 = Fn0; Fc1 = Fn1; Fc2 = Fn2; Fc3 = Fn3;
        Bc0 = Bn0; Bc1 = Bn1; Bc2 = Bn2; Bc3 = Bn3;
        bt0 = nt0; bt1 = nt1; bt2 = nt2; bt3 = nt3;
    }

    // Spill partial 16x64 tile to LDS per ISA C/D layout:
    // VGPR r, lanes 0-15 -> M=r, N=lane ; lanes 16-31 -> M=r+8, N=lane-16
    float* myred = red + wave * 1024;
#pragma unroll
    for (int r = 0; r < 8; ++r) {
        const int row = r + 8 * hv;
        myred[row * 64 +  0 + mrow] = acc0[r];
        myred[row * 64 + 16 + mrow] = acc1[r];
        myred[row * 64 + 32 + mrow] = acc2[r];
        myred[row * 64 + 48 + mrow] = acc3[r];
    }
    __syncthreads();

    // Reduce 8 j-segments, apply ELU, store fp32 (4 straight-line stores).
#pragma unroll
    for (int t = 0; t < 4; ++t) {
        const int idx = tid + t * 256;
        float s = 0.f;
#pragma unroll
        for (int w = 0; w < 8; ++w) s += red[w * 1024 + idx];
        float v = (s > 0.f) ? s : expm1f(s);   // elu, alpha=1
        out[(size_t)(i0 + (idx >> 6)) * OSZ + (idx & 63)] = v;
    }
}

// ---------------------------------------------------------------------------
extern "C" void kernel_launch(void* const* d_in, const int* in_sizes, int n_in,
                              void* d_out, int out_size, void* d_ws, size_t ws_size,
                              hipStream_t stream) {
    const float* x    = (const float*)d_in[0];   // [8192, 256]
    const float* bias = (const float*)d_in[1];   // [8192, 8192]
    const float* W1   = (const float*)d_in[2];   // [64, 256]
    const float* a1   = (const float*)d_in[3];   // [64]
    const float* a2   = (const float*)d_in[4];   // [64]
    float* out = (float*)d_out;                  // [1, 8192, 64]

    char* ws = (char*)d_ws;
    float*  seq = (float*)(ws);                                 // 2 MB
    float*  f1  = (float*)(ws + 2 * 1024 * 1024);               // 32 KB
    float*  f2  = (float*)(ws + 2 * 1024 * 1024 + 32 * 1024);   // 32 KB
    float*  scl = (float*)(ws + 2 * 1024 * 1024 + 64 * 1024);   // 32 KB
    __bf16* sfb = (__bf16*)(ws + 2 * 1024 * 1024 + 96 * 1024);  // 1 MB

    proj_kernel   <<<NN / 4,           256, 0, stream>>>(x, W1, seq);
    fvec_kernel   <<<NN / 256,         256, 0, stream>>>(seq, a1, a2, f1, f2);
    colstat_kernel<<<NN / 256,         256, 0, stream>>>(bias, f1, f2, scl);
    scale_kernel  <<<(NN * OSZ) / 256, 256, 0, stream>>>(seq, scl, sfb);
    attn_kernel   <<<NN / 16,          256, 0, stream>>>(bias, f1, f2, sfb, out);
}
